// EmbeddingRetriever_20727512170512
// MI455X (gfx1250) — matrix operand order repaired and checked by reading, not compiled, not run on hardware
//
#include <hip/hip_runtime.h>
#include <hip/hip_bf16.h>
#include <math.h>

// ---------------------------------------------------------------------------
// MI455X (gfx1250) kNN retrieval.
// HBM-bound problem (205 MB example stream vs 3.3 GFLOP) -> stream examples
// once, hold 32 normalized queries as pre-packed bf16 WMMA A-fragments,
// bf16 hi/lo split (3 WMMA products) for ~fp32 accuracy on the cosine GEMM.
// ---------------------------------------------------------------------------

typedef __bf16 bf16_t;
typedef __attribute__((ext_vector_type(16))) __bf16 v16bf;
typedef __attribute__((ext_vector_type(8)))  __bf16 v8bf;
typedef __attribute__((ext_vector_type(8)))  float  v8f;
typedef __attribute__((ext_vector_type(4)))  float  v4f;

#define NQ      32          // queries
#define NV      2           // vectors per embedding
#define NH      256         // hidden
#define KDIM    512         // NV*NH flattened contraction dim
#define M_EX    100000      // examples
#define NT      64          // examples per block tile
#define LDS_ROW 536         // bf16 elems per LDS row (512 + pad, 16B-mult)
#define TOPK    8

union V16U { v16bf v; v8bf h[2]; };

static __device__ inline v16bf lds_load_v16bf(const bf16_t* p) {
  // 16B-aligned two-half load (row stride 1072B keeps only 16B alignment)
  V16U u;
  u.h[0] = *(const v8bf*)(p);
  u.h[1] = *(const v8bf*)(p + 8);
  return u.v;
}

// ---------------------------------------------------------------------------
// Kernel 1: normalize queries, split fp32 -> bf16 hi/lo, pre-pack into the
// CDNA5 16-bit A-matrix (16x32) lane layout so the GEMM kernel loads
// fragments with a single 32B read per lane.
// wsQ layout: [p(hi/lo)][t(row tile)][c(k-chunk)][lane] * 16 bf16
// ---------------------------------------------------------------------------
__global__ void __launch_bounds__(1024)
qprep_kernel(const float* __restrict__ q, bf16_t* __restrict__ wsQ) {
  __shared__ float s_inv[NQ * NV];
  const int tid = threadIdx.x;

  if (tid < NQ * NV) {                       // one (b,v) row of 256 per thread
    const float* p = q + tid * NH;
    float s = 0.f;
    #pragma unroll 8
    for (int h = 0; h < NH; ++h) { float x = p[h]; s += x * x; }
    s_inv[tid] = 1.0f / fmaxf(sqrtf(s), 1e-12f);
  }
  __syncthreads();

  // tid -> (t, c, L): 2 * 16 * 32 = 1024 fragment-lane entries
  const int t = tid >> 9;
  const int c = (tid >> 5) & 15;
  const int L = tid & 31;
  const int row = t * 16 + (L & 15);
  const int laneHalf = (L >= 16) ? 8 : 0;

  v16bf hi, lo;
  #pragma unroll
  for (int j = 0; j < 16; ++j) {
    const int kk = j + 2 * laneHalf;   // contiguous k map (k = 16h + j), matches B
    const int k  = c * 32 + kk;
    const int v  = k >> 8;
    const float x  = q[row * KDIM + k] * s_inv[row * NV + v];
    const bf16_t h = (bf16_t)x;
    hi[j] = h;
    lo[j] = (bf16_t)(x - (float)h);
  }
  const int idxHi = ((0 * 2 + t) * 16 + c) * 32 + L;
  const int idxLo = ((1 * 2 + t) * 16 + c) * 32 + L;
  *(v16bf*)(wsQ + idxHi * 16) = hi;     // d_ws base is 256B aligned
  *(v16bf*)(wsQ + idxLo * 16) = lo;
}

// ---------------------------------------------------------------------------
// Kernel 2: main GEMM. Each block: 64 examples -> LDS bf16 hi/lo planes,
// 8 waves each compute one 16(query)x16(example) tile with
// v_wmma_f32_16x16x32_bf16, 3 products x 16 k-chunks = 48 WMMA/wave.
// ---------------------------------------------------------------------------
__global__ void __launch_bounds__(256)
dist_kernel(const float* __restrict__ ex, const bf16_t* __restrict__ wsQ,
            float* __restrict__ dist) {
  extern __shared__ __align__(16) char smem_raw[];
  bf16_t* hiP = (bf16_t*)smem_raw;            // [NT][LDS_ROW]
  bf16_t* loP = hiP + NT * LDS_ROW;           // [NT][LDS_ROW]

  const int tid  = threadIdx.x;
  const int lane = tid & 31;
  const int wave = tid >> 5;
  const int m0   = blockIdx.x * NT;

  // ---- stage examples: load fp32, per-(row,v) L2 norm via wave shuffle,
  //      split to bf16 hi/lo, store both planes to LDS ----
  #pragma unroll
  for (int r = 0; r < 8; ++r) {
    const int row = wave * 8 + r;
    int gm = m0 + row; if (gm >= M_EX) gm = M_EX - 1;   // clamp tail (writes guarded later)
    #pragma unroll
    for (int v = 0; v < NV; ++v) {
      const float* p = ex + (size_t)gm * KDIM + v * NH + lane * 8;
      const v4f a = *(const v4f*)p;
      const v4f b = *(const v4f*)(p + 4);
      float s = a[0]*a[0] + a[1]*a[1] + a[2]*a[2] + a[3]*a[3]
              + b[0]*b[0] + b[1]*b[1] + b[2]*b[2] + b[3]*b[3];
      #pragma unroll
      for (int off = 16; off > 0; off >>= 1) s += __shfl_xor(s, off, 32);
      const float inv = 1.0f / fmaxf(sqrtf(s), 1e-12f);

      v8bf h8, l8;
      #pragma unroll
      for (int j = 0; j < 8; ++j) {
        const float x  = ((j < 4) ? a[j] : b[j - 4]) * inv;
        const bf16_t h = (bf16_t)x;
        h8[j] = h;
        l8[j] = (bf16_t)(x - (float)h);
      }
      const int o = row * LDS_ROW + v * NH + lane * 8;
      *(v8bf*)(hiP + o) = h8;
      *(v8bf*)(loP + o) = l8;
    }
  }
  __syncthreads();

  // ---- WMMA: wave -> (t = query tile, nsub = example sub-tile) ----
  const int t       = wave & 1;
  const int nsub    = wave >> 1;
  const int colBase = nsub * 16 + (lane & 15);
  const int kHalf   = (lane >= 16) ? 16 : 0;   // contiguous B-frag k map (k = 16h + s), matches A

  v8f acc = {};
  #pragma unroll 4
  for (int c = 0; c < 16; ++c) {
    const v16bf aHi = *(const v16bf*)(wsQ + (((0 * 2 + t) * 16 + c) * 32 + lane) * 16);
    const v16bf aLo = *(const v16bf*)(wsQ + (((1 * 2 + t) * 16 + c) * 32 + lane) * 16);
    const bf16_t* bp = hiP + colBase * LDS_ROW + c * 32 + kHalf;
    const v16bf bHi = lds_load_v16bf(bp);
    const v16bf bLo = lds_load_v16bf(bp + (size_t)NT * LDS_ROW);
    // cos-sum = qh*eh + qh*el + ql*eh  (ql*el term ~2^-18, dropped)
    acc = __builtin_amdgcn_wmma_f32_16x16x32_bf16(false, aHi, false, bHi, (short)0, acc, false, false);
    acc = __builtin_amdgcn_wmma_f32_16x16x32_bf16(false, aHi, false, bLo, (short)0, acc, false, false);
    acc = __builtin_amdgcn_wmma_f32_16x16x32_bf16(false, aLo, false, bHi, (short)0, acc, false, false);
  }

  // ---- D store: VGPR i holds M = i (+8 for upper lanes), N = lane&15 ----
  const int N     = lane & 15;
  const int mIdx  = m0 + nsub * 16 + N;
  const int mRowO = (lane < 16) ? 0 : 8;
  if (mIdx < M_EX) {
    #pragma unroll
    for (int i = 0; i < 8; ++i) {
      const int b = t * 16 + i + mRowO;
      dist[(size_t)b * M_EX + mIdx] = 1.0f - 0.5f * acc[i];
    }
  }
}

// ---------------------------------------------------------------------------
// Kernel 3: per-query top-8 smallest. Per-thread insertion top-8 over a
// strided scan, then log2(256) LDS tree merge of sorted 8-lists.
// ---------------------------------------------------------------------------
__global__ void __launch_bounds__(256)
topk_kernel(const float* __restrict__ dist, float* __restrict__ out) {
  __shared__ float sd[256][TOPK];
  __shared__ int   si[256][TOPK];
  const int b   = blockIdx.x;
  const int tid = threadIdx.x;

  float bd[TOPK]; int bi[TOPK];
  #pragma unroll
  for (int j = 0; j < TOPK; ++j) { bd[j] = 3.402823e38f; bi[j] = 0x7fffffff; }

  const float* row = dist + (size_t)b * M_EX;
  for (int m = tid; m < M_EX; m += 256) {
    const float d = row[m];
    if (d < bd[TOPK - 1]) {
      int pos = TOPK - 1;
      while (pos > 0 && bd[pos - 1] > d) {
        bd[pos] = bd[pos - 1]; bi[pos] = bi[pos - 1]; --pos;
      }
      bd[pos] = d; bi[pos] = m;
    }
  }
  #pragma unroll
  for (int j = 0; j < TOPK; ++j) { sd[tid][j] = bd[j]; si[tid][j] = bi[j]; }
  __syncthreads();

  for (int s = 128; s > 0; s >>= 1) {
    if (tid < s) {
      float md[TOPK]; int mi[TOPK];
      int ia = 0, ib = 0;
      #pragma unroll
      for (int j = 0; j < TOPK; ++j) {
        bool takeA;
        if (ia >= TOPK)      takeA = false;
        else if (ib >= TOPK) takeA = true;
        else {
          const float da = sd[tid][ia], db = sd[tid + s][ib];
          takeA = (da < db) || (da == db && si[tid][ia] <= si[tid + s][ib]);
        }
        if (takeA) { md[j] = sd[tid][ia];     mi[j] = si[tid][ia];     ++ia; }
        else       { md[j] = sd[tid + s][ib]; mi[j] = si[tid + s][ib]; ++ib; }
      }
      #pragma unroll
      for (int j = 0; j < TOPK; ++j) { sd[tid][j] = md[j]; si[tid][j] = mi[j]; }
    }
    __syncthreads();
  }

  if (tid == 0) {
    #pragma unroll
    for (int j = 0; j < TOPK; ++j) {
      out[b * TOPK + j]             = sd[0][j];          // distances (32,8)
      out[NQ * TOPK + b * TOPK + j] = (float)si[0][j];   // indices   (32,8)
    }
  }
}

// ---------------------------------------------------------------------------
extern "C" void kernel_launch(void* const* d_in, const int* in_sizes, int n_in,
                              void* d_out, int out_size, void* d_ws, size_t ws_size,
                              hipStream_t stream) {
  const float* q  = (const float*)d_in[0];   // (32, 2, 256) fp32
  const float* ex = (const float*)d_in[1];   // (100000, 2, 256) fp32
  // d_in[2] is n == 8 (compile-time TOPK)

  bf16_t* wsQ  = (bf16_t*)d_ws;                          // 64 KB A-fragments
  float*  dist = (float*)((char*)d_ws + 65536);          // 32*100000 fp32
  float*  out  = (float*)d_out;                          // 256 dists + 256 idx

  qprep_kernel<<<1, 1024, 0, stream>>>(q, wsQ);

  const int nblocks = (M_EX + NT - 1) / NT;              // 1563
  const size_t lds  = (size_t)2 * NT * LDS_ROW * sizeof(bf16_t);  // ~134 KB
  dist_kernel<<<nblocks, 256, lds, stream>>>(ex, wsQ, dist);

  topk_kernel<<<NQ, 256, 0, stream>>>(dist, out);
}
